// ModelBlock_704374637202
// MI455X (gfx1250) — matrix operand order repaired and checked by reading, not compiled, not run on hardware
//
#include <hip/hip_runtime.h>
#include <hip/hip_bf16.h>

// ---------------------------------------------------------------------------
// Types for CDNA5 WMMA (wave32, 16x16x32 bf16 -> f32)
// ---------------------------------------------------------------------------
typedef __bf16 bf16;
typedef __bf16 bf16x16 __attribute__((ext_vector_type(16)));
typedef __bf16 bf16x8  __attribute__((ext_vector_type(8)));
typedef float  f32x8   __attribute__((ext_vector_type(8)));

union BF16x16 { bf16x16 v; bf16x8 h[2]; };

__device__ __forceinline__ f32x8 wmma_bf16(bf16x16 a, bf16x16 b, f32x8 c) {
  return __builtin_amdgcn_wmma_f32_16x16x32_bf16(false, a, false, b, (short)0, c,
                                                 false, false);
}

// CDNA5 async global->LDS copy (16B per lane), tracked by ASYNCcnt.
__device__ __forceinline__ void async_load_b128(const bf16* gptr, bf16* lptr) {
  // Generic LDS pointers have the aperture in the high 32 bits; low 32 bits are
  // the LDS byte offset.
  unsigned lds = (unsigned)(unsigned long long)lptr;
  asm volatile("global_load_async_to_lds_b128 %0, %1, off"
               :
               : "v"(lds), "v"(gptr)
               : "memory");
}

__device__ __forceinline__ void wait_async0() {
#if __has_builtin(__builtin_amdgcn_s_wait_asynccnt)
  __builtin_amdgcn_s_wait_asynccnt(0);
#else
  asm volatile("s_wait_asynccnt 0" ::: "memory");
#endif
}

// Problem constants
#define CB 4
#define CS 1024
#define CD 1024
#define CH 16
#define CDH 64
#define CDR 32
#define CDC 256
#define CHID 512
#define CE 16
#define CT (CB * CS)            // 4096 tokens
#define CQD 96                  // DH + DR

// ---------------------------------------------------------------------------
// Epilogue helper: mode 0: C=acc  1: Cbf=acc(bf16)  2: C+=acc
//                  3: C += rowscale[row]*acc
// ---------------------------------------------------------------------------
__device__ __forceinline__ void store_tile(const f32x8& acc, int row0, int col0,
                                           float* __restrict__ C,
                                           bf16* __restrict__ Cbf, int ldc,
                                           int mode,
                                           const float* __restrict__ rowscale,
                                           int rsStride, int l15, int hi) {
#pragma unroll
  for (int r = 0; r < 8; ++r) {
    const int row = row0 + r + hi * 8;
    const size_t idx = (size_t)row * ldc + col0 + l15;
    const float v = acc[r];
    if (mode == 0)      C[idx] = v;
    else if (mode == 1) Cbf[idx] = (bf16)v;
    else if (mode == 2) C[idx] += v;
    else                C[idx] += rowscale[(size_t)row * rsStride] * v;
  }
}

// ---------------------------------------------------------------------------
// LDS-tiled WMMA GEMM:  C[M,N] = A[M,K] * Bt[N,K]^T  (Bt row-major N x K)
// 256 threads / 8 waves per block; block tile 64(M) x 128(N); each wave owns a
// 32x32 sub-tile (4 accumulators).  K consumed in 32-wide panels staged into
// LDS with global_load_async_to_lds_b128 (ASYNCcnt).
// Requires M%64==0, N%128==0, K%32==0.
// ---------------------------------------------------------------------------
__global__ void __launch_bounds__(256)
gemm_wmma_bf16_lds(const bf16* __restrict__ A, const bf16* __restrict__ Bt,
                   float* __restrict__ C, bf16* __restrict__ Cbf,
                   int Kd, int lda, int ldb, int ldc,
                   int mode, const float* __restrict__ rowscale, int rsStride) {
  __shared__ __align__(32) bf16 sA[64 * 32];    // 4KB
  __shared__ __align__(32) bf16 sB[128 * 32];   // 8KB
  const int t = threadIdx.x;
  const int lane = t & 31;
  const int wid = t >> 5;
  const int wm = wid >> 2;         // 0..1  (M sub-block of 32)
  const int wn = wid & 3;          // 0..3  (N sub-block of 32)
  const int l15 = lane & 15;
  const int hi  = (lane >> 4) & 1;
  const int aoff = hi ? 16 : 0;
  const int boff = hi ? 16 : 0;
  const int tm = blockIdx.y, tn = blockIdx.x;

  // Staging assignment: 16B chunk per thread: A = 64 rows x 4 chunks,
  // B = 128 rows x 4 chunks (2 chunks per thread).
  const int srow = t >> 2;
  const int sch  = (t & 3) * 8;
  const bf16* Ag  = A  + (size_t)(tm * 64 + srow) * lda + sch;
  const bf16* Bg0 = Bt + (size_t)(tn * 128 + srow) * ldb + sch;
  const bf16* Bg1 = Bg0 + (size_t)64 * ldb;
  bf16* sAp  = sA + srow * 32 + sch;
  bf16* sBp0 = sB + srow * 32 + sch;
  bf16* sBp1 = sBp0 + 64 * 32;

  f32x8 z = {0.f, 0.f, 0.f, 0.f, 0.f, 0.f, 0.f, 0.f};
  f32x8 acc00 = z, acc01 = z, acc10 = z, acc11 = z;

  for (int k = 0; k < Kd; k += 32) {
    async_load_b128(Ag + k,  sAp);
    async_load_b128(Bg0 + k, sBp0);
    async_load_b128(Bg1 + k, sBp1);
    wait_async0();
    __syncthreads();

    BF16x16 a0, a1;
    const bf16* aBase = sA + (wm * 32 + l15) * 32;
    a0.h[0] = *(const bf16x8*)(aBase + aoff);
    a0.h[1] = *(const bf16x8*)(aBase + aoff + 8);
    a1.h[0] = *(const bf16x8*)(aBase + 16 * 32 + aoff);
    a1.h[1] = *(const bf16x8*)(aBase + 16 * 32 + aoff + 8);
    const bf16* bBase = sB + (wn * 32 + l15) * 32;
    bf16x16 b0 = *(const bf16x16*)(bBase + boff);
    bf16x16 b1 = *(const bf16x16*)(bBase + 16 * 32 + boff);

    acc00 = wmma_bf16(a0.v, b0, acc00);
    acc01 = wmma_bf16(a0.v, b1, acc01);
    acc10 = wmma_bf16(a1.v, b0, acc10);
    acc11 = wmma_bf16(a1.v, b1, acc11);
    __syncthreads();
  }

  const int baseRow = tm * 64 + wm * 32;
  const int baseCol = tn * 128 + wn * 32;
  store_tile(acc00, baseRow,      baseCol,      C, Cbf, ldc, mode, rowscale, rsStride, l15, hi);
  store_tile(acc01, baseRow,      baseCol + 16, C, Cbf, ldc, mode, rowscale, rsStride, l15, hi);
  store_tile(acc10, baseRow + 16, baseCol,      C, Cbf, ldc, mode, rowscale, rsStride, l15, hi);
  store_tile(acc11, baseRow + 16, baseCol + 16, C, Cbf, ldc, mode, rowscale, rsStride, l15, hi);
}

// ---------------------------------------------------------------------------
// Fallback single-wave GEMM for small N (N multiple of 16 only).
// ---------------------------------------------------------------------------
__global__ void __launch_bounds__(32)
gemm_wmma_bf16(const bf16* __restrict__ A, const bf16* __restrict__ Bt,
               float* __restrict__ C, bf16* __restrict__ Cbf,
               int Kd, int lda, int ldb, int ldc,
               int mode, const float* __restrict__ rowscale, int rsStride) {
  const int lane = threadIdx.x & 31;
  const int tn = blockIdx.x, tm = blockIdx.y;
  const int l15 = lane & 15;
  const int hi  = (lane >> 4) & 1;
  const int aoff = hi ? 16 : 0;
  const int boff = hi ? 16 : 0;

  const bf16* Ap = A  + (size_t)(tm * 16 + l15) * lda;
  const bf16* Bp = Bt + (size_t)(tn * 16 + l15) * ldb;

  f32x8 acc = {0.f, 0.f, 0.f, 0.f, 0.f, 0.f, 0.f, 0.f};
  for (int k = 0; k < Kd; k += 32) {
    BF16x16 a;
    a.h[0] = *(const bf16x8*)(Ap + k + aoff);
    a.h[1] = *(const bf16x8*)(Ap + k + aoff + 8);
    bf16x16 b = *(const bf16x16*)(Bp + k + boff);
    acc = wmma_bf16(a.v, b, acc);
  }
  store_tile(acc, tm * 16, tn * 16, C, Cbf, ldc, mode, rowscale, rsStride, l15, hi);
}

// ---------------------------------------------------------------------------
// Elementwise / layout kernels
// ---------------------------------------------------------------------------
__global__ void f32_to_bf16_k(const float* __restrict__ in, bf16* __restrict__ o,
                              int n) {
  int i = blockIdx.x * blockDim.x + threadIdx.x;
  if (i < n) o[i] = (bf16)in[i];
}

// W: R x N row-major f32  ->  Wt: N x R row-major bf16 (i.e. W^T)
__global__ void transpose_f32_to_bf16(const float* __restrict__ W,
                                      bf16* __restrict__ Wt, int R, int N) {
  int i = blockIdx.x * blockDim.x + threadIdx.x;
  if (i >= R * N) return;
  int r = i / N, n = i % N;
  Wt[(size_t)n * R + r] = (bf16)W[i];
}

// q (f32, [T][1536]) -> qpack (bf16, [b][h][s][96]) with RoPE on last 32 and
// 1/sqrt(96) folded in.
__global__ void __launch_bounds__(96)
pack_q(const float* __restrict__ q, bf16* __restrict__ qp) {
  const int s = blockIdx.x, h = blockIdx.y, b = blockIdx.z, j = threadIdx.x;
  const float scale = 0.1020620726f; // 1/sqrt(96)
  const float* qr = q + ((size_t)(b * CS + s)) * (CH * CQD) + h * CQD;
  float val;
  if (j < CDH) {
    val = qr[j];
  } else {
    const int d = j - CDH;
    const int i = d & 15;
    const float fr = __powf(10000.f, -(float)i / 16.f);
    const float ang = (float)s * fr;
    const float c = __cosf(ang), sn = __sinf(ang);
    const float t1 = qr[CDH + i], t2 = qr[CDH + 16 + i];
    val = (d < 16) ? (t1 * c - t2 * sn) : (t1 * sn + t2 * c);
  }
  qp[(((size_t)(b * CH + h)) * CS + s) * CQD + j] = (bf16)(val * scale);
}

// kn_bf ([T][1024]) + kr (f32 [T][32], rope applied here) -> kpack [b][h][s][96]
__global__ void __launch_bounds__(96)
pack_k(const bf16* __restrict__ knbf, const float* __restrict__ kr,
       bf16* __restrict__ kp) {
  const int s = blockIdx.x, h = blockIdx.y, b = blockIdx.z, j = threadIdx.x;
  bf16 out;
  if (j < CDH) {
    out = knbf[((size_t)(b * CS + s)) * (CH * CDH) + h * CDH + j];
  } else {
    const int d = j - CDH;
    const int i = d & 15;
    const float fr = __powf(10000.f, -(float)i / 16.f);
    const float ang = (float)s * fr;
    const float c = __cosf(ang), sn = __sinf(ang);
    const float* krow = kr + ((size_t)(b * CS + s)) * CDR;
    const float t1 = krow[i], t2 = krow[16 + i];
    out = (bf16)((d < 16) ? (t1 * c - t2 * sn) : (t1 * sn + t2 * c));
  }
  kp[(((size_t)(b * CH + h)) * CS + s) * CQD + j] = out;
}

// v_bf ([b][s][h*64+dh]) -> vT ([b][h][dh][s])
__global__ void pack_vT(const bf16* __restrict__ v, bf16* __restrict__ vt) {
  unsigned i = blockIdx.x * blockDim.x + threadIdx.x;
  if (i >= (unsigned)(CT * CH * CDH)) return;
  const unsigned dh = i & 63u;
  const unsigned h  = (i >> 6) & 15u;
  const unsigned bs = i >> 10;
  const unsigned s  = bs & 1023u;
  const unsigned b  = bs >> 10;
  vt[(((size_t)(b * CH + h)) * CDH + dh) * CS + s] = v[i];
}

// ---------------------------------------------------------------------------
// Fused causal attention: one wave handles a 16-query tile for one (b,h).
// 32-key blocks: 6 WMMAs for scores (2 N-tiles x 3 K-steps over 96 dims),
// online softmax, P re-laid out through LDS, 4 WMMAs for P*V.
// ---------------------------------------------------------------------------
__global__ void __launch_bounds__(32)
attn_fwd(const bf16* __restrict__ qp, const bf16* __restrict__ kp,
         const bf16* __restrict__ vt, bf16* __restrict__ O) {
  __shared__ __align__(32) bf16 pT[16 * 32];
  const int lane = threadIdx.x & 31;
  const int qt = blockIdx.x, h = blockIdx.y, b = blockIdx.z;
  const int l15 = lane & 15;
  const int hi  = (lane >> 4) & 1;
  const int aoff = hi ? 16 : 0;
  const int boff = hi ? 16 : 0;
  const size_t bh = (size_t)(b * CH + h);

  const bf16* Qrow = qp + (bh * CS + qt * 16 + l15) * CQD;
  BF16x16 qa[3];
#pragma unroll
  for (int kk = 0; kk < 3; ++kk) {
    qa[kk].h[0] = *(const bf16x8*)(Qrow + kk * 32 + aoff);
    qa[kk].h[1] = *(const bf16x8*)(Qrow + kk * 32 + aoff + 8);
  }

  f32x8 o0 = {0,0,0,0,0,0,0,0}, o1 = o0, o2 = o0, o3 = o0;
  float rmax[8], rsum[8];
#pragma unroll
  for (int r = 0; r < 8; ++r) { rmax[r] = -3.0e38f; rsum[r] = 0.f; }

  const int nkeys = qt * 16 + 16;
  for (int j0 = 0; j0 < nkeys; j0 += 32) {
    f32x8 s0 = {0,0,0,0,0,0,0,0}, s1 = s0;
    const bf16* K0 = kp + (bh * CS + j0 + l15) * CQD;
    const bf16* K1 = K0 + 16 * CQD;
#pragma unroll
    for (int kk = 0; kk < 3; ++kk) {
      bf16x16 bk = *(const bf16x16*)(K0 + kk * 32 + boff);
      s0 = wmma_bf16(qa[kk].v, bk, s0);
    }
#pragma unroll
    for (int kk = 0; kk < 3; ++kk) {
      bf16x16 bk = *(const bf16x16*)(K1 + kk * 32 + boff);
      s1 = wmma_bf16(qa[kk].v, bk, s1);
    }

    const int key0 = j0 + l15, key1 = key0 + 16;
#pragma unroll
    for (int r = 0; r < 8; ++r) {
      const int qrow = qt * 16 + r + hi * 8;
      float v0 = (key0 <= qrow) ? s0[r] : -1.0e9f;
      float v1 = (key1 <= qrow) ? s1[r] : -1.0e9f;
      float m = fmaxf(v0, v1);
      for (int msk = 1; msk < 16; msk <<= 1) m = fmaxf(m, __shfl_xor(m, msk, 32));
      const float nm = fmaxf(rmax[r], m);
      const float corr = __expf(rmax[r] - nm);
      const float p0 = __expf(v0 - nm);
      const float p1 = __expf(v1 - nm);
      float ps = p0 + p1;
      for (int msk = 1; msk < 16; msk <<= 1) ps += __shfl_xor(ps, msk, 32);
      rsum[r] = rsum[r] * corr + ps;
      rmax[r] = nm;
      o0[r] *= corr; o1[r] *= corr; o2[r] *= corr; o3[r] *= corr;
      pT[(r + hi * 8) * 32 + l15]      = (bf16)p0;
      pT[(r + hi * 8) * 32 + 16 + l15] = (bf16)p1;
    }
    __syncthreads();
    BF16x16 pa;
    pa.h[0] = *(const bf16x8*)(&pT[l15 * 32 + aoff]);
    pa.h[1] = *(const bf16x8*)(&pT[l15 * 32 + aoff + 8]);
    const bf16* V0 = vt + (bh * CDH + l15) * CS + j0 + boff;
    o0 = wmma_bf16(pa.v, *(const bf16x16*)(V0),           o0);
    o1 = wmma_bf16(pa.v, *(const bf16x16*)(V0 + 16 * CS), o1);
    o2 = wmma_bf16(pa.v, *(const bf16x16*)(V0 + 32 * CS), o2);
    o3 = wmma_bf16(pa.v, *(const bf16x16*)(V0 + 48 * CS), o3);
    __syncthreads();
  }

#pragma unroll
  for (int r = 0; r < 8; ++r) {
    const int qrow = qt * 16 + r + hi * 8;
    const float inv = 1.0f / rsum[r];
    bf16* orow = O + ((size_t)(b * CS + qrow)) * (CH * CDH) + h * CDH;
    orow[0 * 16 + l15] = (bf16)(o0[r] * inv);
    orow[1 * 16 + l15] = (bf16)(o1[r] * inv);
    orow[2 * 16 + l15] = (bf16)(o2[r] * inv);
    orow[3 * 16 + l15] = (bf16)(o3[r] * inv);
  }
}

// ---------------------------------------------------------------------------
// Residual-add + LayerNorm over D=1024 per row; optional bf16 mirror output.
// ---------------------------------------------------------------------------
__global__ void __launch_bounds__(256)
add_ln(const float* __restrict__ X, const float* __restrict__ Y,
       const float* __restrict__ g, const float* __restrict__ bb,
       float* __restrict__ outf, bf16* __restrict__ outbf, int writeBf) {
  const int row = blockIdx.x, tid = threadIdx.x;
  const float* xr = X + (size_t)row * CD;
  const float* yr = Y + (size_t)row * CD;
  float vals[4];
  float s = 0.f;
#pragma unroll
  for (int i = 0; i < 4; ++i) { vals[i] = xr[tid + i * 256] + yr[tid + i * 256]; s += vals[i]; }
  __shared__ float red[8];
  __shared__ float sMean, sVar;
  for (int m = 16; m > 0; m >>= 1) s += __shfl_down(s, m, 32);
  if ((tid & 31) == 0) red[tid >> 5] = s;
  __syncthreads();
  if (tid == 0) { float t = 0.f; for (int i = 0; i < 8; ++i) t += red[i]; sMean = t / (float)CD; }
  __syncthreads();
  const float mean = sMean;
  float vs = 0.f;
#pragma unroll
  for (int i = 0; i < 4; ++i) { const float d = vals[i] - mean; vs += d * d; }
  __syncthreads();
  for (int m = 16; m > 0; m >>= 1) vs += __shfl_down(vs, m, 32);
  if ((tid & 31) == 0) red[tid >> 5] = vs;
  __syncthreads();
  if (tid == 0) { float t = 0.f; for (int i = 0; i < 8; ++i) t += red[i]; sVar = t / (float)CD; }
  __syncthreads();
  const float inv = rsqrtf(sVar + 1e-5f);
#pragma unroll
  for (int i = 0; i < 4; ++i) {
    const int c = tid + i * 256;
    const float hv = (vals[i] - mean) * inv * g[c] + bb[c];
    outf[(size_t)row * CD + c] = hv;
    if (writeBf) outbf[(size_t)row * CD + c] = (bf16)hv;
  }
}

// silu(a) * b -> bf16
__global__ void silu_mul(const float* __restrict__ a, const float* __restrict__ b3,
                         bf16* __restrict__ o, int n) {
  int i = blockIdx.x * blockDim.x + threadIdx.x;
  if (i >= n) return;
  const float x = a[i];
  const float s = x / (1.f + __expf(-x));
  o[i] = (bf16)(s * b3[i]);
}

// Router: softmax over 16 logits, top-2 (first-index tie-break like jax top_k).
__global__ void router_k(const float* __restrict__ logits, float* __restrict__ probs,
                         float* __restrict__ wfull, int* __restrict__ topi) {
  const int t = blockIdx.x * blockDim.x + threadIdx.x;
  if (t >= CT) return;
  const float* lr = logits + (size_t)t * CE;
  float p[CE];
  float mx = -3.0e38f;
  for (int e = 0; e < CE; ++e) mx = fmaxf(mx, lr[e]);
  float sum = 0.f;
  for (int e = 0; e < CE; ++e) { p[e] = __expf(lr[e] - mx); sum += p[e]; }
  const float is = 1.f / sum;
  for (int e = 0; e < CE; ++e) { p[e] *= is; probs[(size_t)t * CE + e] = p[e]; }
  int i0 = 0;
  for (int e = 1; e < CE; ++e) if (p[e] > p[i0]) i0 = e;
  int i1 = -1;
  for (int e = 0; e < CE; ++e) if (e != i0 && (i1 < 0 || p[e] > p[i1])) i1 = e;
  const float ws = p[i0] + p[i1];
  float* wr = wfull + (size_t)t * CE;
  for (int e = 0; e < CE; ++e) wr[e] = 0.f;
  wr[i0] = p[i0] / ws;
  wr[i1] = p[i1] / ws;
  topi[t * 2 + 0] = i0;
  topi[t * 2 + 1] = i1;
}

// Deterministic per-expert reductions.
__global__ void __launch_bounds__(256)
moe_stats(const float* __restrict__ probs, const int* __restrict__ topi,
          float* __restrict__ counts, float* __restrict__ probsum) {
  const int e = blockIdx.x, tid = threadIdx.x;
  float ps = 0.f, cs = 0.f;
  for (int t = tid; t < CT; t += 256) {
    ps += probs[(size_t)t * CE + e];
    cs += (float)((topi[t * 2 + 0] == e) + (topi[t * 2 + 1] == e));
  }
  __shared__ float sp[256], sc[256];
  sp[tid] = ps; sc[tid] = cs;
  __syncthreads();
  for (int s2 = 128; s2 > 0; s2 >>= 1) {
    if (tid < s2) { sp[tid] += sp[tid + s2]; sc[tid] += sc[tid + s2]; }
    __syncthreads();
  }
  if (tid == 0) { probsum[e] = sp[0]; counts[e] = sc[0]; }
}

__global__ void lose_k(const float* __restrict__ counts,
                       const float* __restrict__ probsum, float* __restrict__ out) {
  if (threadIdx.x == 0 && blockIdx.x == 0) {
    float l = 0.f;
    for (int e = 0; e < CE; ++e)
      l += (counts[e] / ((float)CT * 2.f)) * (probsum[e] / (float)CT);
    out[0] = (float)CE * l;
  }
}

// ---------------------------------------------------------------------------
// Host-side orchestration
// ---------------------------------------------------------------------------
extern "C" void kernel_launch(void* const* d_in, const int* in_sizes, int n_in,
                              void* d_out, int out_size, void* d_ws, size_t ws_size,
                              hipStream_t stream) {
  const float* x    = (const float*)d_in[0];
  // d_in[1] = mask (tril) — causal mask computed analytically in-kernel.
  const float* Wq   = (const float*)d_in[2];
  const float* Wdkv = (const float*)d_in[3];
  const float* Wuk  = (const float*)d_in[4];
  const float* Wuv  = (const float*)d_in[5];
  const float* Wkr  = (const float*)d_in[6];
  const float* Wo   = (const float*)d_in[7];
  const float* g1   = (const float*)d_in[8];
  const float* b1   = (const float*)d_in[9];
  const float* g2   = (const float*)d_in[10];
  const float* b2   = (const float*)d_in[11];
  const float* Wg   = (const float*)d_in[12];
  const float* W1   = (const float*)d_in[13];
  const float* W3   = (const float*)d_in[14];
  const float* W2   = (const float*)d_in[15];
  const float* W1s  = (const float*)d_in[16];
  const float* W3s  = (const float*)d_in[17];
  const float* W2s  = (const float*)d_in[18];
  float* out = (float*)d_out;

  // ---- bump allocator over workspace
  char* ws = (char*)d_ws;
  size_t off = 0;
  auto alloc = [&](size_t bytes) -> void* {
    void* p = ws + off;
    off += (bytes + 255) & ~(size_t)255;
    return p;
  };
  auto abf = [&](size_t n) -> bf16* { return (bf16*)alloc(n * sizeof(bf16)); };
  auto af  = [&](size_t n) -> float* { return (float*)alloc(n * sizeof(float)); };

  bf16* xbf   = abf((size_t)CT * CD);
  bf16* WqT   = abf((size_t)(CH * CQD) * CD);        // 1536 x 1024
  bf16* WdkvT = abf((size_t)CDC * CD);               // 256 x 1024
  bf16* WukT  = abf((size_t)(CH * CDH) * CDC);       // 1024 x 256
  bf16* WuvT  = abf((size_t)(CH * CDH) * CDC);
  bf16* WkrT  = abf((size_t)CDR * CD);               // 32 x 1024
  bf16* WoT   = abf((size_t)CD * (CH * CDH));        // 1024 x 1024
  bf16* WgT   = abf((size_t)CE * CD);                // 16 x 1024
  bf16* W1T   = abf((size_t)CE * CHID * CD);         // per-e: 512 x 1024
  bf16* W3T   = abf((size_t)CE * CHID * CD);
  bf16* W2T   = abf((size_t)CE * CD * CHID);         // per-e: 1024 x 512
  bf16* W1sT  = abf((size_t)CD * CD);
  bf16* W3sT  = abf((size_t)CD * CD);
  bf16* W2sT  = abf((size_t)CD * CD);
  bf16* cbf   = abf((size_t)CT * CDC);
  bf16* knbf  = abf((size_t)CT * CH * CDH);
  bf16* vbf   = abf((size_t)CT * CH * CDH);
  bf16* qpk   = abf((size_t)CB * CH * CS * CQD);
  bf16* kpk   = abf((size_t)CB * CH * CS * CQD);
  bf16* vT    = abf((size_t)CB * CH * CDH * CS);
  bf16* obf   = abf((size_t)CT * CH * CDH);
  bf16* hbf   = abf((size_t)CT * CD);
  bf16* hprod = abf((size_t)CT * CHID);
  bf16* hspr  = abf((size_t)CT * CD);

  float* qf     = af((size_t)CT * CH * CQD);
  float* krf    = af((size_t)CT * CDR);
  float* attprj = af((size_t)CT * CD);
  float* hf     = af((size_t)CT * CD);
  float* glog   = af((size_t)CT * CE);
  float* probs  = af((size_t)CT * CE);
  float* wfull  = af((size_t)CT * CE);
  int*   topi   = (int*)alloc((size_t)CT * 2 * sizeof(int));
  float* he1    = af((size_t)CT * CHID);
  float* he3    = af((size_t)CT * CHID);
  float* hs1    = af((size_t)CT * CD);
  float* hs3    = af((size_t)CT * CD);
  float* ffn    = af((size_t)CT * CD);
  float* counts = af(16);
  float* psum   = af(16);
  (void)ws_size; (void)in_sizes; (void)n_in; (void)out_size;

  auto tr = [&](const float* W, bf16* Wt, int R, int N) {
    int tot = R * N;
    transpose_f32_to_bf16<<<(tot + 255) / 256, 256, 0, stream>>>(W, Wt, R, N);
  };
  auto gemm = [&](const bf16* A, const bf16* Bt, float* C, bf16* Cbf, int N, int Kd,
                  int lda, int ldb, int ldc, int mode, const float* rs, int rss) {
    if ((N & 127) == 0) {
      dim3 grid(N / 128, CT / 64);
      gemm_wmma_bf16_lds<<<grid, 256, 0, stream>>>(A, Bt, C, Cbf, Kd, lda, ldb,
                                                   ldc, mode, rs, rss);
    } else {
      dim3 grid(N / 16, CT / 16);
      gemm_wmma_bf16<<<grid, 32, 0, stream>>>(A, Bt, C, Cbf, Kd, lda, ldb, ldc,
                                              mode, rs, rss);
    }
  };

  // ---- weight transposition + activation conversion (bf16)
  f32_to_bf16_k<<<(CT * CD + 255) / 256, 256, 0, stream>>>(x, xbf, CT * CD);
  tr(Wq,   WqT,   CD, CH * CQD);
  tr(Wdkv, WdkvT, CD, CDC);
  tr(Wuk,  WukT,  CDC, CH * CDH);
  tr(Wuv,  WuvT,  CDC, CH * CDH);
  tr(Wkr,  WkrT,  CD, CDR);
  tr(Wo,   WoT,   CH * CDH, CD);
  tr(Wg,   WgT,   CD, CE);
  for (int e = 0; e < CE; ++e) {
    tr(W1 + (size_t)e * CD * CHID, W1T + (size_t)e * CHID * CD, CD, CHID);
    tr(W3 + (size_t)e * CD * CHID, W3T + (size_t)e * CHID * CD, CD, CHID);
    tr(W2 + (size_t)e * CHID * CD, W2T + (size_t)e * CD * CHID, CHID, CD);
  }
  tr(W1s, W1sT, CD, CD);
  tr(W3s, W3sT, CD, CD);
  tr(W2s, W2sT, CD, CD);

  // ---- projections
  gemm(xbf, WqT,   qf,  nullptr, CH * CQD, CD,  CD,  CD,  CH * CQD, 0, nullptr, 0);
  gemm(xbf, WdkvT, nullptr, cbf, CDC,      CD,  CD,  CD,  CDC,      1, nullptr, 0);
  gemm(cbf, WukT,  nullptr, knbf, CH * CDH, CDC, CDC, CDC, CH * CDH, 1, nullptr, 0);
  gemm(cbf, WuvT,  nullptr, vbf,  CH * CDH, CDC, CDC, CDC, CH * CDH, 1, nullptr, 0);
  gemm(xbf, WkrT,  krf, nullptr, CDR,      CD,  CD,  CD,  CDR,      0, nullptr, 0);

  // ---- pack Q/K (rope folded), V transpose
  {
    dim3 g(CS, CH, CB);
    pack_q<<<g, 96, 0, stream>>>(qf, qpk);
    pack_k<<<g, 96, 0, stream>>>(knbf, krf, kpk);
    pack_vT<<<(CT * CH * CDH + 255) / 256, 256, 0, stream>>>(vbf, vT);
  }

  // ---- fused causal attention
  {
    dim3 g(CS / 16, CH, CB);
    attn_fwd<<<g, 32, 0, stream>>>(qpk, kpk, vT, obf);
  }

  // ---- output projection + LN1
  gemm(obf, WoT, attprj, nullptr, CD, CH * CDH, CH * CDH, CH * CDH, CD, 0, nullptr, 0);
  add_ln<<<CT, 256, 0, stream>>>(x, attprj, g1, b1, hf, hbf, 1);

  // ---- gating / routing
  gemm(hbf, WgT, glog, nullptr, CE, CD, CD, CD, CE, 0, nullptr, 0);
  router_k<<<(CT + 255) / 256, 256, 0, stream>>>(glog, probs, wfull, topi);
  moe_stats<<<CE, 256, 0, stream>>>(probs, topi, counts, psum);

  // ---- shared FFN (initializes ffn buffer)
  gemm(hbf, W1sT, hs1, nullptr, CD, CD, CD, CD, CD, 0, nullptr, 0);
  gemm(hbf, W3sT, hs3, nullptr, CD, CD, CD, CD, CD, 0, nullptr, 0);
  silu_mul<<<(CT * CD + 255) / 256, 256, 0, stream>>>(hs1, hs3, hspr, CT * CD);
  gemm(hspr, W2sT, ffn, nullptr, CD, CD, CD, CD, CD, 0, nullptr, 0);

  // ---- routed experts: ffn += wfull[:,e] * (silu(h@W1e) * (h@W3e)) @ W2e
  for (int e = 0; e < CE; ++e) {
    const bf16* w1t = W1T + (size_t)e * CHID * CD;
    const bf16* w3t = W3T + (size_t)e * CHID * CD;
    const bf16* w2t = W2T + (size_t)e * CD * CHID;
    gemm(hbf, w1t, he1, nullptr, CHID, CD, CD, CD, CHID, 0, nullptr, 0);
    gemm(hbf, w3t, he3, nullptr, CHID, CD, CD, CD, CHID, 0, nullptr, 0);
    silu_mul<<<(CT * CHID + 255) / 256, 256, 0, stream>>>(he1, he3, hprod, CT * CHID);
    gemm(hprod, w2t, ffn, nullptr, CD, CHID, CHID, CHID, CD, 3, wfull + e, CE);
  }

  // ---- final LN -> out, aux loss -> out[T*D]
  add_ln<<<CT, 256, 0, stream>>>(hf, ffn, g2, b2, out, nullptr, 0);
  lose_k<<<1, 32, 0, stream>>>(counts, psum, out + (size_t)CT * CD);
}
